// MultiPerspective_71201967833843
// MI455X (gfx1250) — matrix operand order repaired and checked by reading, not compiled, not run on hardware
//
#include <hip/hip_runtime.h>
#include <hip/hip_bf16.h>

#define DEV __device__ __forceinline__

namespace {
constexpr int kB = 16, kL = 128, kH = 256, kP = 20;
constexpr float kEPS = 1e-8f;
constexpr int kOut = 8 * kP; // 160 output features per (b,l)
}

typedef __bf16 bf16_t;
typedef __attribute__((ext_vector_type(16))) __bf16 v16bf;
typedef __attribute__((ext_vector_type(8)))  __bf16 v8bf;
typedef __attribute__((ext_vector_type(8)))  float  v8f;

// Guarded fast division: x / max(d, EPS) via v_rcp_f32 (reference clamps with
// EPS anyway; correctly-rounded IEEE quotient is unnecessary here).
DEV float div_eps(float x, float d) {
  return x * __builtin_amdgcn_rcpf(d > kEPS ? d : kEPS);
}

// A fragment, 16x32 bf16, row-major source. Lane l<16 -> row l, K chunks [k0+0..7]
// and [k0+16..23]; lane l>=16 -> row l-16, K chunks [k0+8..15] and [k0+24..31].
DEV v16bf load_a_frag(const bf16_t* __restrict__ row, int k0, int hi) {
  v8bf lo = *(const v8bf*)(row + k0 + 8 * hi);
  v8bf hh = *(const v8bf*)(row + k0 + 16 + 8 * hi);
  v16bf a;
#pragma unroll
  for (int e = 0; e < 8; ++e) { a[e] = lo[e]; a[8 + e] = hh[e]; }
  return a;
}

// B fragment, 32x16 bf16. Lane l holds column l%16; lanes 0-15 K chunks
// [k0+0..7] and [k0+16..23], lanes 16-31 K chunks [k0+8..15] and [k0+24..31].
// colptr points at the contiguous column (= row of the row-major transposed operand).
DEV v16bf load_b_frag(const bf16_t* __restrict__ col, int k0, int hi) {
  v8bf lo = *(const v8bf*)(col + k0 + 8 * hi);
  v8bf hh = *(const v8bf*)(col + k0 + 16 + 8 * hi);
  v16bf b;
#pragma unroll
  for (int e = 0; e < 8; ++e) { b[e] = lo[e]; b[8 + e] = hh[e]; }
  return b;
}

DEV v8f wmma_bf16(v16bf a, v16bf b, v8f c) {
  return __builtin_amdgcn_wmma_f32_16x16x32_bf16(false, a, false, b, (short)0, c,
                                                 false, false);
}

// ---------------------------------------------------------------- K1: convert
// One block per (dir,b,i) row: bf16 copies of v1/v2, transposed v2 copy, norms.
__global__ void k_convert(const float* __restrict__ h1, const float* __restrict__ h2,
                          bf16_t* __restrict__ v1h, bf16_t* __restrict__ v2h,
                          bf16_t* __restrict__ v2hT,
                          float* __restrict__ v1norm, float* __restrict__ v2norm) {
  int blk = blockIdx.x;                 // [0, 2*B*L)
  int dir = blk / (kB * kL);
  int rem = blk % (kB * kL);
  int b = rem / kL, i = rem % kL;
  int t = threadIdx.x;                  // h in [0,256)
  long rowdb = (long)(dir * kB + b) * kL + i;
  float x1 = h1[((long)(b * kL + i)) * (2 * kH) + dir * kH + t];
  float x2 = h2[((long)(b * kL + i)) * (2 * kH) + dir * kH + t];
  v1h[rowdb * kH + t] = (bf16_t)x1;
  v2h[rowdb * kH + t] = (bf16_t)x2;
  v2hT[((long)(dir * kB + b) * kH + t) * kL + i] = (bf16_t)x2;
  __shared__ float red[256];
  red[t] = x1 * x1; __syncthreads();
  for (int s = 128; s > 0; s >>= 1) { if (t < s) red[t] += red[t + s]; __syncthreads(); }
  if (t == 0) v1norm[rowdb] = sqrtf(red[0]);
  __syncthreads();
  red[t] = x2 * x2; __syncthreads();
  for (int s = 128; s > 0; s >>= 1) { if (t < s) red[t] += red[t + s]; __syncthreads(); }
  if (t == 0) v2norm[rowdb] = sqrtf(red[0]);
}

// ---------------------------------------------------------------- K1b: last state
__global__ void k_last(const float* __restrict__ h2, const int* __restrict__ h2_mask,
                       float* __restrict__ h2_last) {
  int blk = blockIdx.x;                 // 2*B
  int dir = blk / kB, b = blk % kB;
  int t = threadIdx.x;                  // h
  int idx = 0;
  if (dir == 0) {
    int len = 0;
    for (int j = 0; j < kL; ++j) len += h2_mask[b * kL + j];
    idx = len - 1; if (idx < 0) idx = 0; if (idx > kL - 1) idx = kL - 1;
  }
  h2_last[(long)(dir * kB + b) * kH + t] =
      h2[((long)(b * kL + idx)) * (2 * kH) + dir * kH + t];
}

// ---------------------------------------------------------------- K2: attention scores
// One block per (dir,b): S = v1 @ v2^T (128x128, K=256) via WMMA bf16, fused
// cosine normalization -> attn (f32 + bf16 copies).
__global__ void k_scores(const bf16_t* __restrict__ v1h, const bf16_t* __restrict__ v2h,
                         const float* __restrict__ v1norm, const float* __restrict__ v2norm,
                         float* __restrict__ attnS, bf16_t* __restrict__ attnH) {
  int blk = blockIdx.x;                 // 2*B
  long base = (long)blk * kL;
  const bf16_t* v1p = v1h + base * kH;
  const bf16_t* v2p = v2h + base * kH;
  int wave = threadIdx.x >> 5, lane = threadIdx.x & 31;
  int r = lane & 15, hi = lane >> 4;
  int itile = wave;
  const bf16_t* arow = v1p + (long)(itile * 16 + r) * kH;
  v16bf afr[8];
#pragma unroll
  for (int kt = 0; kt < 8; ++kt) afr[kt] = load_a_frag(arow, 32 * kt, hi);
  float n1[8];
#pragma unroll
  for (int vr = 0; vr < 8; ++vr) n1[vr] = v1norm[base + itile * 16 + vr + 8 * hi];
  for (int jt = 0; jt < 8; ++jt) {
    v8f acc = {};
    const bf16_t* bcol = v2p + (long)(jt * 16 + r) * kH;
#pragma unroll
    for (int kt = 0; kt < 8; ++kt)
      acc = wmma_bf16(afr[kt], load_b_frag(bcol, 32 * kt, hi), acc);
    int j = jt * 16 + r;
    float n2 = v2norm[base + j];
#pragma unroll
    for (int vr = 0; vr < 8; ++vr) {
      int i = itile * 16 + vr + 8 * hi;
      float a = div_eps(acc[vr], n1[vr] * n2);
      attnS[(base + i) * kL + j] = a;
      attnH[(base + i) * kL + j] = (bf16_t)a;
    }
  }
}

// ---------------------------------------------------------------- K2b: attn row-sum + max-product
__global__ void k_attn_aux(const float* __restrict__ attnS, const float* __restrict__ h2,
                           float* __restrict__ attn_sum, float* __restrict__ attn_hmax) {
  int blk = blockIdx.x;                 // 2*B*L
  int dir = blk / (kB * kL);
  int rem = blk % (kB * kL);
  int b = rem / kL, i = rem % kL;
  int t = threadIdx.x;                  // h in [0,256)
  long rowdb = (long)(dir * kB + b) * kL + i;
  __shared__ float sa[kL];
  if (t < kL) sa[t] = attnS[rowdb * kL + t];
  __syncthreads();
  if (t == 0) {
    float s = 0.f;
    for (int j = 0; j < kL; ++j) s += sa[j];
    attn_sum[rowdb] = s;
  }
  float mx = -3.402823466e38f;
  const float* v2col = h2 + (long)b * kL * (2 * kH) + dir * kH + t;
  for (int j = 0; j < kL; ++j)
    mx = fmaxf(mx, sa[j] * v2col[(long)j * (2 * kH)]);
  attn_hmax[rowdb * kH + t] = mx;
}

// ---------------------------------------------------------------- K3: attn_mean GEMM
// One block per (dir,b): attn_mean = (attn @ v2) / attn_sum  (128x256, K=128).
__global__ void k_attn_mean(const bf16_t* __restrict__ attnH, const bf16_t* __restrict__ v2hT,
                            const float* __restrict__ attn_sum,
                            float* __restrict__ attn_mean) {
  int blk = blockIdx.x;                 // 2*B
  long base = (long)blk * kL;
  const bf16_t* ap = attnH + base * kL;
  const bf16_t* btp = v2hT + (long)blk * kH * kL;
  int wave = threadIdx.x >> 5, lane = threadIdx.x & 31;
  int r = lane & 15, hi = lane >> 4;
  int itile = wave;
  const bf16_t* arow = ap + (long)(itile * 16 + r) * kL;
  v16bf afr[4];
#pragma unroll
  for (int kt = 0; kt < 4; ++kt) afr[kt] = load_a_frag(arow, 32 * kt, hi);
  float sinv[8];
#pragma unroll
  for (int vr = 0; vr < 8; ++vr) {
    float s = attn_sum[base + itile * 16 + vr + 8 * hi];
    sinv[vr] = __builtin_amdgcn_rcpf(s > kEPS ? s : kEPS);
  }
  for (int nt = 0; nt < 16; ++nt) {
    v8f acc = {};
    const bf16_t* bcol = btp + (long)(nt * 16 + r) * kL;
#pragma unroll
    for (int kt = 0; kt < 4; ++kt)
      acc = wmma_bf16(afr[kt], load_b_frag(bcol, 32 * kt, hi), acc);
    int hcol = nt * 16 + r;
#pragma unroll
    for (int vr = 0; vr < 8; ++vr) {
      int i = itile * 16 + vr + 8 * hi;
      attn_mean[(base + i) * kH + hcol] = acc[vr] * sinv[vr];
    }
  }
}

// ---------------------------------------------------------------- K4: max-pooling matching
// One block per (dir,b,p): n_p = (v1 .* w2[p]) @ v2^T, fused cosine norm + row max.
__global__ void k_maxpool(const bf16_t* __restrict__ v1h, const bf16_t* __restrict__ v2h,
                          const float* __restrict__ fw2, const float* __restrict__ bw2,
                          float* __restrict__ out) {
  int dir = blockIdx.x, b = blockIdx.y, p = blockIdx.z;
  long base = (long)(dir * kB + b) * kL;
  const bf16_t* v1p = v1h + base * kH;
  const bf16_t* v2p = v2h + base * kH;
  const float* w = (dir ? bw2 : fw2) + (long)p * kH;
  int t = threadIdx.x;
  __shared__ float s_w2[kH];
  __shared__ float s_n1[kL];
  __shared__ float s_n2[kL];
  s_w2[t] = w[t] * w[t];
  __syncthreads();
  {
    int row = t & 127;
    const bf16_t* vp = (t < 128) ? (v1p + (long)row * kH) : (v2p + (long)row * kH);
    float s = 0.f;
    for (int h = 0; h < kH; ++h) { float a = (float)vp[h]; s += a * a * s_w2[h]; }
    if (t < 128) s_n1[row] = sqrtf(s); else s_n2[row] = sqrtf(s);
  }
  __syncthreads();
  int wave = t >> 5, lane = t & 31;
  int r = lane & 15, hi = lane >> 4;
  int itile = wave;
  const bf16_t* arow = v1p + (long)(itile * 16 + r) * kH;
  v16bf afr[8];
#pragma unroll
  for (int kt = 0; kt < 8; ++kt) {
    int k0 = 32 * kt;
#pragma unroll
    for (int e = 0; e < 8; ++e) {
      int hA = k0 + 8 * hi + e;
      int hB = k0 + 16 + 8 * hi + e;
      afr[kt][e]     = (bf16_t)((float)arow[hA] * s_w2[hA]);
      afr[kt][8 + e] = (bf16_t)((float)arow[hB] * s_w2[hB]);
    }
  }
  float mx[8];
#pragma unroll
  for (int vr = 0; vr < 8; ++vr) mx[vr] = -3.402823466e38f;
  for (int jt = 0; jt < 8; ++jt) {
    v8f acc = {};
    const bf16_t* bcol = v2p + (long)(jt * 16 + r) * kH;
#pragma unroll
    for (int kt = 0; kt < 8; ++kt)
      acc = wmma_bf16(afr[kt], load_b_frag(bcol, 32 * kt, hi), acc);
    int j = jt * 16 + r;
    float n2 = s_n2[j];
#pragma unroll
    for (int vr = 0; vr < 8; ++vr) {
      int i = itile * 16 + vr + 8 * hi;
      float m = div_eps(acc[vr], s_n1[i] * n2);
      mx[vr] = fmaxf(mx[vr], m);
    }
  }
  // max over the 16 lanes of each half-group (same rows, different columns)
#pragma unroll
  for (int mask = 1; mask <= 8; mask <<= 1)
#pragma unroll
    for (int vr = 0; vr < 8; ++vr)
      mx[vr] = fmaxf(mx[vr], __shfl_xor(mx[vr], mask, 32));
  if (r == 0) {
#pragma unroll
    for (int vr = 0; vr < 8; ++vr) {
      int i = itile * 16 + vr + 8 * hi;
      out[((long)(b * kL) + i) * kOut + dir * 80 + 20 + p] = mx[vr];
    }
  }
}

// ---------------------------------------------------------------- K5: full matchings (VALU)
__global__ void k_fullmatch(const float* __restrict__ h1,
                            const float* __restrict__ h2_last,
                            const float* __restrict__ attn_mean,
                            const float* __restrict__ attn_hmax,
                            const float* __restrict__ fw1, const float* __restrict__ bw1,
                            const float* __restrict__ fw3, const float* __restrict__ bw3,
                            const float* __restrict__ fw4, const float* __restrict__ bw4,
                            float* __restrict__ out) {
  int blk = blockIdx.x;                 // 2*B*L
  int dir = blk / (kB * kL);
  int rem = blk % (kB * kL);
  int b = rem / kL, i = rem % kL;
  int t = threadIdx.x;
  long rowdb = (long)(dir * kB + b) * kL + i;
  __shared__ float s_v1[kH], s_last[kH], s_mean[kH], s_hmax[kH];
  s_v1[t]   = h1[((long)(b * kL + i)) * (2 * kH) + dir * kH + t];
  s_last[t] = h2_last[(long)(dir * kB + b) * kH + t];
  s_mean[t] = attn_mean[rowdb * kH + t];
  s_hmax[t] = attn_hmax[rowdb * kH + t];
  __syncthreads();
  if (t < 3 * kP) {
    int kind = t / kP, p = t % kP;      // 0: full, 1: att(m1), 2: matt(m2)
    const float* w = (kind == 0) ? (dir ? bw1 : fw1)
                   : (kind == 1) ? (dir ? bw3 : fw3)
                                 : (dir ? bw4 : fw4);
    w += (long)p * kH;
    const float* v2v = (kind == 0) ? s_last : (kind == 1) ? s_mean : s_hmax;
    float num = 0.f, sA = 0.f, sC = 0.f;
    for (int h = 0; h < kH; ++h) {
      float wv = w[h]; float w2 = wv * wv;
      float a = s_v1[h], c = v2v[h];
      num += a * c * w2; sA += a * a * w2; sC += c * c * w2;
    }
    float d = fmaxf(sqrtf(sA), kEPS) * fmaxf(sqrtf(sC), kEPS);
    int off = (kind == 0) ? 0 : (kind == 1) ? 40 : 60;
    out[((long)(b * kL) + i) * kOut + dir * 80 + off + p] =
        num * __builtin_amdgcn_rcpf(d);
  }
}

extern "C" void kernel_launch(void* const* d_in, const int* in_sizes, int n_in,
                              void* d_out, int out_size, void* d_ws, size_t ws_size,
                              hipStream_t stream) {
  (void)in_sizes; (void)n_in; (void)out_size; (void)ws_size;
  const float* h1 = (const float*)d_in[0];
  const float* h2 = (const float*)d_in[1];
  const int* h2_mask = (const int*)d_in[3];
  const float* fw1 = (const float*)d_in[4];
  const float* bw1 = (const float*)d_in[5];
  const float* fw2 = (const float*)d_in[6];
  const float* bw2 = (const float*)d_in[7];
  const float* fw3 = (const float*)d_in[8];
  const float* bw3 = (const float*)d_in[9];
  const float* fw4 = (const float*)d_in[10];
  const float* bw4 = (const float*)d_in[11];
  float* out = (float*)d_out;

  char* ws = (char*)d_ws;
  size_t off = 0;
  auto take = [&](size_t bytes) -> char* {
    char* pp = ws + off;
    off = (off + bytes + 255) & ~(size_t)255;
    return pp;
  };
  const size_t nRows = (size_t)2 * kB * kL;   // 4096 (dir,b,row)
  bf16_t* v1h   = (bf16_t*)take(nRows * kH * sizeof(bf16_t));
  bf16_t* v2h   = (bf16_t*)take(nRows * kH * sizeof(bf16_t));
  bf16_t* v2hT  = (bf16_t*)take((size_t)2 * kB * kH * kL * sizeof(bf16_t));
  float* v1norm = (float*)take(nRows * sizeof(float));
  float* v2norm = (float*)take(nRows * sizeof(float));
  float* attnS  = (float*)take(nRows * kL * sizeof(float));
  bf16_t* attnH = (bf16_t*)take(nRows * kL * sizeof(bf16_t));
  float* attn_sum  = (float*)take(nRows * sizeof(float));
  float* attn_mean = (float*)take(nRows * kH * sizeof(float));
  float* attn_hmax = (float*)take(nRows * kH * sizeof(float));
  float* h2_last   = (float*)take((size_t)2 * kB * kH * sizeof(float));

  k_convert<<<2 * kB * kL, 256, 0, stream>>>(h1, h2, v1h, v2h, v2hT, v1norm, v2norm);
  k_last<<<2 * kB, 256, 0, stream>>>(h2, h2_mask, h2_last);
  k_scores<<<2 * kB, 256, 0, stream>>>(v1h, v2h, v1norm, v2norm, attnS, attnH);
  k_attn_aux<<<2 * kB * kL, 256, 0, stream>>>(attnS, h2, attn_sum, attn_hmax);
  k_attn_mean<<<2 * kB, 256, 0, stream>>>(attnH, v2hT, attn_sum, attn_mean);
  dim3 g4(2, kB, kP);
  k_maxpool<<<g4, 256, 0, stream>>>(v1h, v2h, fw2, bw2, out);
  k_fullmatch<<<2 * kB * kL, 256, 0, stream>>>(h1, h2_last, attn_mean, attn_hmax,
                                               fw1, bw1, fw3, bw3, fw4, bw4, out);
}
